// FlashSVDRoPEAttention_85529978733231
// MI455X (gfx1250) — matrix operand order repaired and checked by reading, not compiled, not run on hardware
//
#include <hip/hip_runtime.h>
#include <hip/hip_bf16.h>

typedef __attribute__((ext_vector_type(16))) _Float16 v16h;
typedef __attribute__((ext_vector_type(8)))  _Float16 v8h;
typedef __attribute__((ext_vector_type(8)))  float    v8f;

#define Bsz 4
#define Hn  16
#define Mn  2048
#define Rn  32
#define DH  64
#define NEGF (-1e30f)
#define SCALE 0.125f   // 1/sqrt(64)

union F8  { v8f v; float f[8]; };
union V16 { v16h v; v8h h[2]; _Float16 e[16]; };

// ---- CDNA5 async global->LDS copy (ASYNCcnt path, ISA §15.18.3 op 98) -----
__device__ __forceinline__ void async_ld_b128(unsigned lds_addr, int voff_bytes,
                                              const void* sbase) {
    // GVS form: vdst = LDS byte address, voff = per-lane byte offset,
    // s[base:base+1] = uniform 64-bit base.
    asm volatile("global_load_async_to_lds_b128 %0, %1, %2"
                 :: "v"(lds_addr), "v"(voff_bytes), "s"(sbase) : "memory");
}
__device__ __forceinline__ void wait_async0() {
#if __has_builtin(__builtin_amdgcn_s_wait_asynccnt)
    __builtin_amdgcn_s_wait_asynccnt(0);
#else
    asm volatile("s_wait_asynccnt 0x0" ::: "memory");
#endif
}

// ---------------------------------------------------------------------------
// Kernel 1: Q/K/V = P @ V + bias, RoPE(Q,K), emit f16.
// One thread per (b,h,m,p) with p in [0,32): handles dims d=p and d=p+32.
// ---------------------------------------------------------------------------
__global__ __launch_bounds__(256) void qkv_rope_kernel(
    const float* __restrict__ Pq, const float* __restrict__ Pk, const float* __restrict__ Pv,
    const float* __restrict__ Vq, const float* __restrict__ Vk, const float* __restrict__ Vv,
    const float* __restrict__ bq, const float* __restrict__ bk, const float* __restrict__ bv,
    const int*   __restrict__ pos,
    _Float16* __restrict__ Qh, _Float16* __restrict__ Kh, _Float16* __restrict__ Vh)
{
    const long idx = (long)blockIdx.x * 256 + threadIdx.x;   // B*H*M*32 threads
    const int  p   = (int)(idx & 31);
    const long m   = (idx >> 5) % Mn;
    const long h   = ((idx >> 5) / Mn) % Hn;
    const long b   = (idx >> 5) / ((long)Mn * Hn);

    const long  prow = ((b * Hn + h) * Mn + m) * Rn;
    const float* pq = Pq + prow;
    const float* pk = Pk + prow;
    const float* pv = Pv + prow;
    const float* vq = Vq + h * (Rn * DH);
    const float* vk = Vk + h * (Rn * DH);
    const float* vv = Vv + h * (Rn * DH);

    float q1 = 0.f, q2 = 0.f, k1 = 0.f, k2 = 0.f, v1 = 0.f, v2 = 0.f;
#pragma unroll 8
    for (int r = 0; r < Rn; ++r) {
        const float aq = pq[r], ak = pk[r], av = pv[r];
        q1 = fmaf(aq, vq[r * DH + p],      q1);
        q2 = fmaf(aq, vq[r * DH + p + 32], q2);
        k1 = fmaf(ak, vk[r * DH + p],      k1);
        k2 = fmaf(ak, vk[r * DH + p + 32], k2);
        v1 = fmaf(av, vv[r * DH + p],      v1);
        v2 = fmaf(av, vv[r * DH + p + 32], v2);
    }
    q1 += bq[h * DH + p]; q2 += bq[h * DH + p + 32];
    k1 += bk[h * DH + p]; k2 += bk[h * DH + p + 32];
    v1 += bv[h * DH + p]; v2 += bv[h * DH + p + 32];

    const float pp   = (float)pos[b * Mn + m];
    const float freq = __powf(10000.0f, -(float)(p & ~1) * (1.0f / 64.0f));
    const float ang  = pp * freq;
    const float c = __cosf(ang), s = __sinf(ang);

    const float qr1 = q1 * c - q2 * s, qr2 = q1 * s + q2 * c;
    const float kr1 = k1 * c - k2 * s, kr2 = k1 * s + k2 * c;

    const long o = ((b * Hn + h) * Mn + m) * DH + p;
    Qh[o] = (_Float16)qr1; Qh[o + 32] = (_Float16)qr2;
    Kh[o] = (_Float16)kr1; Kh[o + 32] = (_Float16)kr2;
    Vh[o] = (_Float16)v1;  Vh[o + 32] = (_Float16)v2;
}

// ---------------------------------------------------------------------------
// Kernel 2: flash attention. 8 waves/block, 16 query rows per wave,
// double-buffered 32-key LDS tiles (K via async DMA), WMMA f16->f32.
// ---------------------------------------------------------------------------
__global__ __launch_bounds__(256) void flash_attn_kernel(
    const _Float16* __restrict__ Qh, const _Float16* __restrict__ Kh,
    const _Float16* __restrict__ Vh, const int* __restrict__ amask,
    float* __restrict__ out)
{
    const int mtile = blockIdx.x;       // M/128 tiles
    const int h     = blockIdx.y;
    const int b     = blockIdx.z;
    const int wave  = threadIdx.x >> 5;
    const int lane  = threadIdx.x & 31;
    const int half  = lane >> 4;        // 0: lanes 0-15, 1: lanes 16-31
    const int ln    = lane & 15;

    const int  mbase = mtile * 128;
    const int  qrow0 = mbase + wave * 16;
    const long bh    = (long)b * Hn + h;

    __shared__ __align__(32) _Float16 Ks [2][32][DH];   // K tiles (double buf)
    __shared__ __align__(32) _Float16 Vts[2][DH][32];   // V^T tiles (double buf)
    __shared__ __align__(32) _Float16 Ps [8][16][32];   // per-wave P staging

    const _Float16* Kbh = Kh + bh * Mn * DH;
    const _Float16* Vbh = Vh + bh * Mn * DH;

    // ---- load Q A-fragments once: rows qrow0+ln, d-chunks [0,32) and [32,64)
    const _Float16* Qp = Qh + (bh * Mn + qrow0 + ln) * DH;
    V16 aQ[2];
#pragma unroll
    for (int c = 0; c < 2; ++c) {
        aQ[c].h[0] = *(const v8h*)(Qp + 32 * c + 16 * half);
        aQ[c].h[1] = *(const v8h*)(Qp + 32 * c + 16 * half + 8);
    }

    F8 acc[4];
#pragma unroll
    for (int t = 0; t < 4; ++t) { v8f z = {}; acc[t].v = z; }
    float mrow[8], lrow[8];
#pragma unroll
    for (int j = 0; j < 8; ++j) { mrow[j] = -3.0e38f; lrow[j] = 0.f; }

    const int kend = mbase + 128;          // causal extent of this block
    const int srow = (int)threadIdx.x >> 3;          // 0..31 tile row
    const int scol = ((int)threadIdx.x & 7) * 8;     // 0..56 tile col (halves)

    // ---- prologue: fill buffer 0 (K async, V through VGPRs + transpose)
    {
        const int voff = (srow * DH + scol) * (int)sizeof(_Float16);
        async_ld_b128((unsigned)(unsigned long long)&Ks[0][srow][scol], voff, Kbh);
        _Float16 tmp[8];
        *(v8h*)tmp = *(const v8h*)(Vbh + srow * DH + scol);
#pragma unroll
        for (int i = 0; i < 8; ++i) Vts[0][scol + i][srow] = tmp[i];
    }

    for (int key0 = 0; key0 < kend; key0 += 32) {
        const int cur = (key0 >> 5) & 1;
        const int nxt = cur ^ 1;
        const int knext = key0 + 32;
        const bool have_next = knext < kend;

        wait_async0();       // async fill of `cur` has landed in LDS
        __syncthreads();     // + all waves' V stores / prior reads drained

        // ---- kick off next tile: K via async DMA, V global loads hoisted
        if (have_next) {
            const int voff = ((knext + srow) * DH + scol) * (int)sizeof(_Float16);
            async_ld_b128((unsigned)(unsigned long long)&Ks[nxt][srow][scol],
                          voff, Kbh);
        }
        _Float16 vtmp[8];
        if (have_next) {
            *(v8h*)vtmp = *(const v8h*)(Vbh + (knext + srow) * DH + scol);
            __builtin_prefetch(Vbh + (knext + 32 + srow) * DH + scol, 0, 1);
        }

        if (key0 <= qrow0 + 15) {   // wave-uniform causal skip
            // ---- S = Q K^T : two 16-key tiles, contraction over DH in 2 chunks
            F8 s[2];
#pragma unroll
            for (int t = 0; t < 2; ++t) {
                v8f z = {}; s[t].v = z;
#pragma unroll
                for (int c = 0; c < 2; ++c) {
                    v16h bK = *(const v16h*)&Ks[cur][16 * t + ln][32 * c + 16 * half];
                    s[t].v = __builtin_amdgcn_wmma_f32_16x16x32_f16(
                        false, aQ[c].v, false, bK, (short)0, s[t].v, false, false);
                }
            }

            // ---- online softmax over this 32-key block
            const int am0 = amask[b * Mn + key0 + ln];
            const int am1 = amask[b * Mn + key0 + 16 + ln];
            const int k0  = key0 + ln, k1 = key0 + 16 + ln;
#pragma unroll
            for (int j = 0; j < 8; ++j) {
                const int mr = qrow0 + j + 8 * half;
                float x0 = (k0 <= mr && am0) ? s[0].f[j] * SCALE : NEGF;
                float x1 = (k1 <= mr && am1) ? s[1].f[j] * SCALE : NEGF;
                float v = fmaxf(x0, x1);
                v = fmaxf(v, __shfl_xor(v, 1, 16));
                v = fmaxf(v, __shfl_xor(v, 2, 16));
                v = fmaxf(v, __shfl_xor(v, 4, 16));
                v = fmaxf(v, __shfl_xor(v, 8, 16));
                const float mn = fmaxf(mrow[j], v);
                const float sf = __expf(mrow[j] - mn);
                mrow[j] = mn;
                const float p0 = __expf(x0 - mn);
                const float p1 = __expf(x1 - mn);
                float rs = p0 + p1;
                rs += __shfl_xor(rs, 1, 16);
                rs += __shfl_xor(rs, 2, 16);
                rs += __shfl_xor(rs, 4, 16);
                rs += __shfl_xor(rs, 8, 16);
                lrow[j] = lrow[j] * sf + rs;
#pragma unroll
                for (int t = 0; t < 4; ++t) acc[t].f[j] *= sf;
                Ps[wave][j + 8 * half][ln]      = (_Float16)p0;
                Ps[wave][j + 8 * half][16 + ln] = (_Float16)p1;
            }

            // ---- O += P V
            V16 aP;
            aP.h[0] = *(const v8h*)&Ps[wave][ln][16 * half];
            aP.h[1] = *(const v8h*)&Ps[wave][ln][16 * half + 8];
#pragma unroll
            for (int t = 0; t < 4; ++t) {
                v16h bV = *(const v16h*)&Vts[cur][16 * t + ln][16 * half];
                acc[t].v = __builtin_amdgcn_wmma_f32_16x16x32_f16(
                    false, aP.v, false, bV, (short)0, acc[t].v, false, false);
            }
        }

        // ---- finish next V tile: transposed store into the other buffer
        if (have_next) {
#pragma unroll
            for (int i = 0; i < 8; ++i) Vts[nxt][scol + i][srow] = vtmp[i];
        }
    }

    // ---- normalize and write O in (B, M, H, DH) order
#pragma unroll
    for (int j = 0; j < 8; ++j) {
        const int mr = qrow0 + j + 8 * half;
        const float inv = lrow[j] > 0.f ? 1.0f / lrow[j] : 0.f;
        float* op = out + (((long)b * Mn + mr) * Hn + h) * DH;
#pragma unroll
        for (int t = 0; t < 4; ++t) op[16 * t + ln] = acc[t].f[j] * inv;
    }
}

// ---------------------------------------------------------------------------
extern "C" void kernel_launch(void* const* d_in, const int* in_sizes, int n_in,
                              void* d_out, int out_size, void* d_ws, size_t ws_size,
                              hipStream_t stream)
{
    const float* Pq = (const float*)d_in[0];
    const float* Pk = (const float*)d_in[1];
    const float* Pv = (const float*)d_in[2];
    const float* Vq = (const float*)d_in[3];
    const float* Vk = (const float*)d_in[4];
    const float* Vv = (const float*)d_in[5];
    const float* bq = (const float*)d_in[6];
    const float* bk = (const float*)d_in[7];
    const float* bv = (const float*)d_in[8];
    const int* amask = (const int*)d_in[9];
    const int* pos   = (const int*)d_in[10];
    float* out = (float*)d_out;

    const long NE = (long)Bsz * Hn * Mn * DH;   // 8388608 elements
    _Float16* Qh = (_Float16*)d_ws;
    _Float16* Kh = Qh + NE;
    _Float16* Vh = Kh + NE;

    {
        const long total = (long)Bsz * Hn * Mn * 32;
        dim3 grid((unsigned)(total / 256)), block(256);
        qkv_rope_kernel<<<grid, block, 0, stream>>>(Pq, Pk, Pv, Vq, Vk, Vv,
                                                    bq, bk, bv, pos, Qh, Kh, Vh);
    }
    {
        dim3 grid(Mn / 128, Hn, Bsz), block(256);
        flash_attn_kernel<<<grid, block, 0, stream>>>(Qh, Kh, Vh, amask, out);
    }
}
